// CrystalGraphConvNet_52536039964946
// MI455X (gfx1250) — matrix operand order repaired and checked by reading, not compiled, not run on hardware
//
#include <hip/hip_runtime.h>
#include <hip/hip_bf16.h>
#include <math.h>

// ---------------------------------------------------------------------------
// CGCNN forward for MI455X (gfx1250), wave32.
//  - GEMMs on v_wmma_f32_16x16x32_f16 (f16 mul, f32 acc).
//  - Weights pre-converted to transposed f16 so both A and B fragments are
//    contiguous 2x ds_load_b128 per lane.
//  - W tiles staged global->LDS with GLOBAL_LOAD_ASYNC_TO_LDS_B128 (ASYNCcnt).
//  - BatchNorm batch-stats => 2-pass GEMM (stats pass + apply pass).
// Workspace: 2 * N*F f32 + f16 weights + stats (~31 MB).
// ---------------------------------------------------------------------------

typedef __attribute__((ext_vector_type(16))) _Float16 v16h;
typedef __attribute__((ext_vector_type(8)))  float    v8f;

#define NATOM 60000
#define MNBR  12
#define FDIM  64
#define BDIM  41
#define ORIGD 92
#define HDIM  128
#define TWOF  128
#define CDIM  169          // 2F + B
#define KPAD  192          // CDIM padded to 6 chunks of 32
#define KPADE 96           // ORIGD padded to 3 chunks of 32
#define NCONV 3
#define NROW  (NATOM*MNBR) // 720000, divisible by 128
#define NCRY  1200
#define APC   50
#define EPSBN 1e-5f
#define SLOPE 0.01f
#define ALD   40           // A-tile LDS leading dim (halves): 80B rows, 16B aligned

__device__ __forceinline__ float leakyf(float v)   { return v >= 0.f ? v : SLOPE * v; }
__device__ __forceinline__ float sigmoidf(float v) { return 1.f / (1.f + __expf(-v)); }

__device__ __forceinline__ v8f wmma_f16(v16h a, v16h b, v8f c) {
  return __builtin_amdgcn_wmma_f32_16x16x32_f16(false, a, false, b, (short)0, c,
                                                false, false);
}

// Async global->LDS 16-byte copy (gfx1250 VFLAT op 98, tracked by ASYNCcnt).
// %0 = VGPR holding wave-relative LDS byte address, %1 = 64-bit global address.
__device__ __forceinline__ void async_g2l_b128(unsigned lds_off, const void* g) {
  asm volatile("global_load_async_to_lds_b128 %0, %1, off"
               :: "v"(lds_off), "v"(g) : "memory");
}
__device__ __forceinline__ void wait_async0() {
  asm volatile("s_wait_asynccnt 0" ::: "memory");
}
__device__ __forceinline__ unsigned lds_addr32(const void* p) {
  return (unsigned)(size_t)p;  // low 32 bits of generic ptr == LDS byte address
}

// A fragment (16x32 f16, M x K), LDS layout [row][k] with leading dim ldk.
// lane = 16*half + row; elems 0..7 <- K = 8*half+0..7; 8..15 <- K = 16+8*half+0..7.
__device__ __forceinline__ v16h load_a_frag(const _Float16* base, int ldk) {
  const int lane = threadIdx.x & 31;
  const _Float16* p = base + (lane & 15) * ldk;
  const int h8 = (lane >> 4) * 8;
  v16h a;
#pragma unroll
  for (int e = 0; e < 8; ++e) a[e] = p[h8 + e];
#pragma unroll
  for (int e = 0; e < 8; ++e) a[8 + e] = p[16 + h8 + e];
  return a;
}

// B fragment (32x16 f16, K x N) from TRANSPOSED LDS tile Wt[col][k], ld = 32.
// lane = 16*half + col; elems 0..7 <- K = 8*half+0..7; 8..15 <- K = 16+8*half+0..7.
__device__ __forceinline__ v16h load_b_fragT(const _Float16* Wt, int colBase) {
  const int lane = threadIdx.x & 31;
  const _Float16* p = Wt + (colBase + (lane & 15)) * 32 + (lane >> 4) * 8;
  v16h b;
#pragma unroll
  for (int e = 0; e < 8; ++e) b[e] = p[e];
#pragma unroll
  for (int e = 0; e < 8; ++e) b[8 + e] = p[16 + e];
  return b;
}

// ---------------------------------------------------------------------------
// Weight pre-conversion: f32 row-major -> f16 transposed [col][k], K zero-padded.
// ---------------------------------------------------------------------------
__global__ void conv_w_to_f16(const float* __restrict__ W, _Float16* __restrict__ Wh) {
  int idx = blockIdx.x * 256 + threadIdx.x;         // over NCONV*TWOF*KPAD
  if (idx >= NCONV * TWOF * KPAD) return;
  int l = idx / (TWOF * KPAD);
  int rem = idx - l * (TWOF * KPAD);
  int d = rem / KPAD;
  int k = rem - d * KPAD;
  Wh[idx] = (_Float16)((k < CDIM) ? W[(size_t)l * CDIM * TWOF + k * TWOF + d] : 0.f);
}

__global__ void embed_w_to_f16(const float* __restrict__ W, _Float16* __restrict__ Wh) {
  int idx = blockIdx.x * 256 + threadIdx.x;         // over FDIM*KPADE
  if (idx >= FDIM * KPADE) return;
  int d = idx / KPADE;
  int k = idx - d * KPADE;
  Wh[idx] = (_Float16)((k < ORIGD) ? W[k * FDIM + d] : 0.f);
}

// ---------------------------------------------------------------------------
// Embed: x = atom_fea(N x 92) @ embed_W(92 x 64) + embed_b.
// 256-row x 64-col tile; 8 waves = 2 row-halves x 4 N-tiles.
// ---------------------------------------------------------------------------
__global__ void embed_gemm(const float* __restrict__ atom_fea,
                           const _Float16* __restrict__ WhE,   // [64][KPADE]
                           const float* __restrict__ embed_b,
                           float* __restrict__ xf) {
  __shared__ _Float16 A_s[256 * ALD];
  __shared__ _Float16 Wt_s[64 * 32];
  const int rowBase = blockIdx.x * 256;
  const int wv = threadIdx.x >> 5;
  const int lane = threadIdx.x & 31;
  const int ntile = wv & 3;
  const int rhalf = wv >> 2;

  v8f acc[8];
#pragma unroll
  for (int mt = 0; mt < 8; ++mt)
#pragma unroll
    for (int i = 0; i < 8; ++i) acc[mt][i] = 0.f;

  for (int c = 0; c < 3; ++c) {
    __syncthreads();
    for (int e = threadIdx.x; e < 256 * 32; e += 256) {
      int r = e >> 5, kk = e & 31;
      int kg = c * 32 + kk;
      int row = rowBase + r;
      float v = (row < NATOM && kg < ORIGD) ? atom_fea[(size_t)row * ORIGD + kg] : 0.f;
      A_s[r * ALD + kk] = (_Float16)v;
    }
    {   // async-stage W chunk: 64 cols x 32 halves = 4 KB, 16B per thread
      int q = threadIdx.x;            // 0..255
      int col = q >> 2, seg = q & 3;  // seg = 16B segment of the 64B col-row
      async_g2l_b128(lds_addr32(&Wt_s[col * 32 + seg * 8]),
                     &WhE[(size_t)col * KPADE + c * 32 + seg * 8]);
    }
    wait_async0();
    __syncthreads();
    v16h bf = load_b_fragT(Wt_s, ntile * 16);
#pragma unroll
    for (int mt = 0; mt < 8; ++mt) {
      v16h af = load_a_frag(A_s + (rhalf * 128 + mt * 16) * ALD, ALD);
      acc[mt] = wmma_f16(af, bf, acc[mt]);
    }
  }

  const int col = ntile * 16 + (lane & 15);
  const float bias = embed_b[col];
#pragma unroll
  for (int mt = 0; mt < 8; ++mt)
#pragma unroll
    for (int i = 0; i < 8; ++i) {
      int row = rowBase + rhalf * 128 + mt * 16 + (lane >> 4) * 8 + i;
      if (row < NATOM) xf[(size_t)row * FDIM + col] = acc[mt][i] + bias;
    }
}

// ---------------------------------------------------------------------------
// Conv GEMM body (shared between stats / apply passes).
// 128-row x 128-col tile; wave w owns N-tile w; 8 M-tiles per wave.
// Implicit A row r = [x[n], x[nbr_idx[r]], nbr_fea[r]] (169, padded 192).
// ---------------------------------------------------------------------------
#define CONV_GEMM_BODY(xf, nbr_fea, Wh, A_s, Wt_s, nidx_s, selfn_s, rowBase, acc)   \
  for (int c = 0; c < 6; ++c) {                                                     \
    __syncthreads();                                                                \
    for (int e = threadIdx.x; e < 128 * 32; e += 256) {                             \
      int r = e >> 5, kk = e & 31;                                                  \
      int kg = c * 32 + kk;                                                         \
      float v = 0.f;                                                                \
      if (kg < FDIM)          v = xf[(size_t)selfn_s[r] * FDIM + kg];               \
      else if (kg < 2 * FDIM) v = xf[(size_t)nidx_s[r] * FDIM + (kg - FDIM)];       \
      else if (kg < CDIM)     v = nbr_fea[(size_t)(rowBase + r) * BDIM + (kg - 2 * FDIM)]; \
      A_s[r * ALD + kk] = (_Float16)v;                                              \
    }                                                                               \
    _Pragma("unroll")                                                               \
    for (int e = 0; e < 2; ++e) {  /* 128 cols x 64B = 8KB, 2x16B per thread */     \
      int q = threadIdx.x + e * 256;                                                \
      int col = q >> 2, seg = q & 3;                                                \
      async_g2l_b128(lds_addr32(&Wt_s[col * 32 + seg * 8]),                         \
                     &Wh[(size_t)col * KPAD + c * 32 + seg * 8]);                   \
    }                                                                               \
    wait_async0();                                                                  \
    __syncthreads();                                                                \
    v16h bf = load_b_fragT(Wt_s, wv * 16);                                          \
    _Pragma("unroll")                                                               \
    for (int mt = 0; mt < 8; ++mt) {                                                \
      v16h af = load_a_frag(A_s + mt * 16 * ALD, ALD);                              \
      acc[mt] = wmma_f16(af, bf, acc[mt]);                                          \
    }                                                                               \
  }

// Pass 1: per-channel sum / sumsq of raw GEMM output (BN1 batch stats).
__global__ void conv_stats(const float* __restrict__ xf,
                           const float* __restrict__ nbr_fea,
                           const int* __restrict__ nbr_idx,
                           const _Float16* __restrict__ Wh,  // [128][KPAD] transposed
                           float* __restrict__ stats1) {     // [128 sum | 128 sumsq]
  __shared__ _Float16 A_s[128 * ALD];
  __shared__ _Float16 Wt_s[128 * 32];
  __shared__ int nidx_s[128];
  __shared__ int selfn_s[128];
  const int rowBase = blockIdx.x * 128;
  const int wv = threadIdx.x >> 5;
  const int lane = threadIdx.x & 31;

  if (threadIdx.x < 128) {
    int r = threadIdx.x;
    int nb = nbr_idx[rowBase + r];
    nidx_s[r] = nb;
    selfn_s[r] = (rowBase + r) / MNBR;
    __builtin_prefetch(&xf[(size_t)nb * FDIM], 0, 1);  // global_prefetch_b8
  }

  v8f acc[8];
#pragma unroll
  for (int mt = 0; mt < 8; ++mt)
#pragma unroll
    for (int i = 0; i < 8; ++i) acc[mt][i] = 0.f;

  CONV_GEMM_BODY(xf, nbr_fea, Wh, A_s, Wt_s, nidx_s, selfn_s, rowBase, acc)

  float s1 = 0.f, s2 = 0.f;
#pragma unroll
  for (int mt = 0; mt < 8; ++mt)
#pragma unroll
    for (int i = 0; i < 8; ++i) { float v = acc[mt][i]; s1 += v; s2 += v * v; }
  // lanes L and L+16 hold the same column (different rows) -> fold halves
  s1 += __shfl_xor(s1, 16, 32);
  s2 += __shfl_xor(s2, 16, 32);
  if (lane < 16) {
    int col = wv * 16 + lane;
    atomicAdd(&stats1[col], s1);
    atomicAdd(&stats1[TWOF + col], s2);
  }
}

// Fold BN1 into scale/shift: BN(gemm + conv_b) = gemm*scale + shift,
// scale = g1*rstd, shift = b1 - mean(gemm)*scale (conv bias cancels).
__global__ void bn1_prep(const float* __restrict__ stats1,
                         const float* __restrict__ g1,
                         const float* __restrict__ b1,
                         float* __restrict__ scale1,
                         float* __restrict__ shift1) {
  int d = threadIdx.x;
  if (d < TWOF) {
    float inv = 1.f / (float)NROW;
    float mg = stats1[d] * inv;
    float var = stats1[TWOF + d] * inv - mg * mg;
    float sc = g1[d] * rsqrtf(var + EPSBN);
    scale1[d] = sc;
    shift1[d] = b1[d] - mg * sc;
  }
}

// Pass 2: recompute tiles, apply BN1, sigmoid(filter)*leaky(core),
// atomically accumulate the M-sum into nbr_accum[n, f].
__global__ void conv_apply(const float* __restrict__ xf,
                           const float* __restrict__ nbr_fea,
                           const int* __restrict__ nbr_idx,
                           const _Float16* __restrict__ Wh,
                           const float* __restrict__ scale1,
                           const float* __restrict__ shift1,
                           float* __restrict__ nbr_accum) {
  __shared__ _Float16 A_s[128 * ALD];
  __shared__ _Float16 Wt_s[128 * 32];
  __shared__ int nidx_s[128];
  __shared__ int selfn_s[128];
  __shared__ float core_s[128 * 64];
  const int rowBase = blockIdx.x * 128;
  const int wv = threadIdx.x >> 5;
  const int lane = threadIdx.x & 31;

  if (threadIdx.x < 128) {
    int r = threadIdx.x;
    int nb = nbr_idx[rowBase + r];
    nidx_s[r] = nb;
    selfn_s[r] = (rowBase + r) / MNBR;
    __builtin_prefetch(&xf[(size_t)nb * FDIM], 0, 1);
  }

  v8f acc[8];
#pragma unroll
  for (int mt = 0; mt < 8; ++mt)
#pragma unroll
    for (int i = 0; i < 8; ++i) acc[mt][i] = 0.f;

  CONV_GEMM_BODY(xf, nbr_fea, Wh, A_s, Wt_s, nidx_s, selfn_s, rowBase, acc)

  const int col = wv * 16 + (lane & 15);
  const float sc = scale1[col];
  const float sh = shift1[col];
  __syncthreads();
  if (wv >= 4) {  // core half: leaky(BN(z)), stage for filter waves
#pragma unroll
    for (int mt = 0; mt < 8; ++mt)
#pragma unroll
      for (int i = 0; i < 8; ++i) {
        int row = mt * 16 + (lane >> 4) * 8 + i;
        core_s[row * 64 + (col - 64)] = leakyf(acc[mt][i] * sc + sh);
      }
  }
  __syncthreads();
  if (wv < 4) {   // filter half: sigmoid(BN(z)) * core, m-sum via atomics
#pragma unroll
    for (int mt = 0; mt < 8; ++mt)
#pragma unroll
      for (int i = 0; i < 8; ++i) {
        int row = mt * 16 + (lane >> 4) * 8 + i;
        float s = sigmoidf(acc[mt][i] * sc + sh) * core_s[row * 64 + col];
        atomicAdd(&nbr_accum[(size_t)selfn_s[row] * FDIM + col], s);
      }
  }
}

// BN2 stats over nbr_accum (N x 64).
__global__ void bn2_stats(const float* __restrict__ nbr, float* __restrict__ stats2) {
  int col = threadIdx.x & 63;
  int rstart = blockIdx.x * (blockDim.x >> 6) + (threadIdx.x >> 6);
  int rstep = gridDim.x * (blockDim.x >> 6);
  float s = 0.f, q = 0.f;
  for (int r = rstart; r < NATOM; r += rstep) {
    float v = nbr[(size_t)r * FDIM + col];
    s += v; q += v * v;
  }
  atomicAdd(&stats2[col], s);
  atomicAdd(&stats2[FDIM + col], q);
}

// x = leaky(x + BN2(nbr_sum))
__global__ void bn2_apply(float* __restrict__ xf,
                          const float* __restrict__ nbr,
                          const float* __restrict__ stats2,
                          const float* __restrict__ g2,
                          const float* __restrict__ b2) {
  int idx = blockIdx.x * blockDim.x + threadIdx.x;
  if (idx >= NATOM * FDIM) return;
  int f = idx & 63;
  float inv = 1.f / (float)NATOM;
  float m = stats2[f] * inv;
  float var = stats2[FDIM + f] * inv - m * m;
  float rs = rsqrtf(var + EPSBN);
  float v = xf[idx] + (nbr[idx] - m) * rs * g2[f] + b2[f];
  xf[idx] = leakyf(v);
}

__global__ void zero_f32(float* __restrict__ p, int n) {
  int i = blockIdx.x * blockDim.x + threadIdx.x;
  if (i < n) p[i] = 0.f;
}

// fc->vis has no activation between: collapse to w_eff(64) = fc_W @ vis_W,
// b_eff = fc_b @ vis_W + vis_b.
__global__ void make_weff(const float* __restrict__ fc_W,
                          const float* __restrict__ fc_b,
                          const float* __restrict__ vis_W,
                          const float* __restrict__ vis_b,
                          float* __restrict__ weff) {
  int f = threadIdx.x;
  if (f < FDIM) {
    float a = 0.f;
    for (int h = 0; h < HDIM; ++h) a += fc_W[f * HDIM + h] * vis_W[h];
    weff[f] = a;
  }
  if (f == 0) {
    float a = vis_b[0];
    for (int h = 0; h < HDIM; ++h) a += fc_b[h] * vis_W[h];
    weff[FDIM] = a;
  }
}

__global__ void scalar_vis(const float* __restrict__ xf,
                           const float* __restrict__ weff,
                           const float* __restrict__ conn,
                           float* __restrict__ vis,
                           float* __restrict__ scalar) {
  __shared__ float w_s[FDIM + 1];
  if (threadIdx.x <= FDIM) w_s[threadIdx.x] = weff[threadIdx.x];
  __syncthreads();
  int n = blockIdx.x * blockDim.x + threadIdx.x;
  if (n >= NATOM) return;
  float s = w_s[FDIM];
  for (int f = 0; f < FDIM; ++f) s += xf[(size_t)n * FDIM + f] * w_s[f];
  scalar[n] = s;
  vis[n] = s * conn[n];
}

__global__ void pool_cry(const int* __restrict__ cry,
                         const float* __restrict__ vis,
                         float* __restrict__ out) {
  int c = blockIdx.x * blockDim.x + threadIdx.x;
  if (c >= NCRY) return;
  float s = 0.f, nnz = 0.f;
  for (int a = 0; a < APC; ++a) {
    float v = vis[cry[c * APC + a]];
    s += v;
    nnz += (v != 0.f) ? 1.f : 0.f;
  }
  out[c] = s / nnz;
}

extern "C" void kernel_launch(void* const* d_in, const int* in_sizes, int n_in,
                              void* d_out, int out_size, void* d_ws, size_t ws_size,
                              hipStream_t stream) {
  (void)in_sizes; (void)n_in; (void)out_size; (void)ws_size;
  const float* atom_fea = (const float*)d_in[0];
  const float* nbr_fea  = (const float*)d_in[1];
  const int*   nbr_idx  = (const int*)d_in[2];
  const int*   cry_idx  = (const int*)d_in[3];
  /* d_in[4] distances: unused by reference */
  const float* conn     = (const float*)d_in[5];
  const float* embed_W  = (const float*)d_in[6];
  const float* embed_b  = (const float*)d_in[7];
  const float* conv_W   = (const float*)d_in[8];
  /* d_in[9] conv_b: cancels inside training-mode BN */
  const float* bn1_g    = (const float*)d_in[10];
  const float* bn1_b    = (const float*)d_in[11];
  const float* bn2_g    = (const float*)d_in[12];
  const float* bn2_b    = (const float*)d_in[13];
  const float* fc_W     = (const float*)d_in[14];
  const float* fc_b     = (const float*)d_in[15];
  const float* vis_W    = (const float*)d_in[16];
  const float* vis_b    = (const float*)d_in[17];

  // Output layout: (out[1200], atom_fea_vis[60000], atom_scalar[60000])
  float* out_pool   = (float*)d_out;
  float* out_vis    = out_pool + NCRY;
  float* out_scalar = out_vis + NATOM;

  // Workspace: x, nbr_accum, stats (contiguous for bulk zeroing), f16 weights.
  float* xf     = (float*)d_ws;
  float* nbrac  = xf + (size_t)NATOM * FDIM;
  float* stats1 = nbrac + (size_t)NATOM * FDIM;   // 256 floats
  float* stats2 = stats1 + 2 * TWOF;              // 128 floats
  float* scale1 = stats2 + 2 * FDIM;              // 128
  float* shift1 = scale1 + TWOF;                  // 128
  float* weff   = shift1 + TWOF;                  // 65, padded to 80 (16B align)
  _Float16* WhC = (_Float16*)(weff + 80);         // 3*128*192 halves (transposed)
  _Float16* WhE = WhC + NCONV * TWOF * KPAD;      // 64*96 halves (transposed)

  conv_w_to_f16<<<(NCONV * TWOF * KPAD + 255) / 256, 256, 0, stream>>>(conv_W, WhC);
  embed_w_to_f16<<<(FDIM * KPADE + 255) / 256, 256, 0, stream>>>(embed_W, WhE);

  embed_gemm<<<(NATOM + 255) / 256, 256, 0, stream>>>(atom_fea, WhE, embed_b, xf);

  const int zn = NATOM * FDIM + 2 * TWOF + 2 * FDIM;  // nbrac + stats1 + stats2
  for (int l = 0; l < NCONV; ++l) {
    const _Float16* Wl = WhC + (size_t)l * TWOF * KPAD;
    zero_f32<<<(zn + 255) / 256, 256, 0, stream>>>(nbrac, zn);
    conv_stats<<<NROW / 128, 256, 0, stream>>>(xf, nbr_fea, nbr_idx, Wl, stats1);
    bn1_prep<<<1, TWOF, 0, stream>>>(stats1, bn1_g + l * TWOF, bn1_b + l * TWOF,
                                     scale1, shift1);
    conv_apply<<<NROW / 128, 256, 0, stream>>>(xf, nbr_fea, nbr_idx, Wl,
                                               scale1, shift1, nbrac);
    bn2_stats<<<256, 256, 0, stream>>>(nbrac, stats2);
    bn2_apply<<<(NATOM * FDIM + 255) / 256, 256, 0, stream>>>(
        xf, nbrac, stats2, bn2_g + l * FDIM, bn2_b + l * FDIM);
  }

  make_weff<<<1, 64, 0, stream>>>(fc_W, fc_b, vis_W, vis_b, weff);
  scalar_vis<<<(NATOM + 255) / 256, 256, 0, stream>>>(xf, weff, conn,
                                                      out_vis, out_scalar);
  pool_cry<<<(NCRY + 255) / 256, 256, 0, stream>>>(cry_idx, out_vis, out_pool);
}
